// UNeXt3D_mlp_attention_boundary_51187420234166
// MI455X (gfx1250) — matrix operand order repaired and checked by reading, compiled, not run on hardware
//
#include <hip/hip_runtime.h>
#include <hip/hip_bf16.h>

#define HWD 48
#define NTOK (HWD*HWD*HWD)   // 110592 tokens
#define CCH 128
#define EPSF 1e-5f

typedef __attribute__((ext_vector_type(16))) __bf16 v16bf;
typedef __attribute__((ext_vector_type(8)))  float  v8f;
typedef __attribute__((ext_vector_type(4)))  unsigned int u32x4;
typedef __attribute__((ext_vector_type(2)))  unsigned int u32x2;

union Frag { v16bf v; struct { u32x4 lo, hi; } q; };

__device__ __forceinline__ unsigned short f2bf(float f){
  unsigned int u = __float_as_uint(f);
  u += 0x7FFFu + ((u >> 16) & 1u);       // round-to-nearest-even
  return (unsigned short)(u >> 16);
}
__device__ __forceinline__ float bf2f(unsigned short h){
  return __uint_as_float(((unsigned int)h) << 16);
}

__device__ __forceinline__ float block_sum128(float v, float* red, int tid){
  red[tid] = v; __syncthreads();
  for (int s = 64; s > 0; s >>= 1){ if (tid < s) red[tid] += red[tid + s]; __syncthreads(); }
  float r = red[0]; __syncthreads();
  return r;
}

// ---------------------------------------------------------------------------
// Kernel 1: stride-2 3x3x3 proj conv (4->128) + bias + LN(g0,b0) + LN(g2,b2)
// one block = one token, 128 threads = 128 out channels
// ---------------------------------------------------------------------------
__global__ void proj_ln_kernel(const float* __restrict__ x, const float* __restrict__ pw,
    const float* __restrict__ pb, const float* __restrict__ g0, const float* __restrict__ b0,
    const float* __restrict__ g2, const float* __restrict__ b2,
    float* __restrict__ t_out, unsigned short* __restrict__ h_out)
{
  __shared__ float patch[108];
  __shared__ float red[128];
  const int n = blockIdx.x;
  const int tid = threadIdx.x;
  const int d0 = n % HWD, w0 = (n / HWD) % HWD, h0 = n / (HWD*HWD);
  if (tid < 108){
    int ci = tid / 27, kk = tid % 27;
    int k0 = kk / 9, k1 = (kk % 9) / 3, k2 = kk % 3;
    int ih = 2*h0 - 1 + k0, iw = 2*w0 - 1 + k1, id = 2*d0 - 1 + k2;
    float v = 0.f;
    if ((unsigned)ih < 96u && (unsigned)iw < 96u && (unsigned)id < 96u)
      v = x[ci*884736 + ih*9216 + iw*96 + id];
    patch[tid] = v;
  }
  __syncthreads();
  const float* wr = pw + tid*108;
  float acc = pb[tid];
  #pragma unroll 4
  for (int j = 0; j < 108; ++j) acc += patch[j] * wr[j];

  float m   = block_sum128(acc, red, tid) * (1.f/128.f);
  float dv  = acc - m;
  float var = block_sum128(dv*dv, red, tid) * (1.f/128.f);
  float t   = dv * rsqrtf(var + EPSF) * g0[tid] + b0[tid];

  float m2   = block_sum128(t, red, tid) * (1.f/128.f);
  float dv2  = t - m2;
  float var2 = block_sum128(dv2*dv2, red, tid) * (1.f/128.f);
  float h    = dv2 * rsqrtf(var2 + EPSF) * g2[tid] + b2[tid];

  t_out[(long)n*CCH + tid] = t;
  h_out[(long)n*CCH + tid] = f2bf(h);
}

// ---------------------------------------------------------------------------
// Kernel 2/4/6: fused axial-shift (channel-group spatial roll) + 128x128 GEMM
// (bf16 WMMA, f32 acc) + bias [+ GeLU + residual + final LN + transpose out]
// block = 256 threads (8 waves), 128 tokens; wave = 16 tokens x 128 out-ch
// ---------------------------------------------------------------------------
template<int AXIS, bool FINAL>
__global__ void shift_gemm_kernel(const unsigned short* __restrict__ hin,
    const float* __restrict__ fcw, const float* __restrict__ fcb,
    unsigned short* __restrict__ hout,
    const float* __restrict__ tres, const float* __restrict__ gn,
    const float* __restrict__ bn, float* __restrict__ outp)
{
  extern __shared__ char smem[];
  unsigned short* Alds = (unsigned short*)smem;            // 128x128 bf16 (32 KB)
  unsigned short* Blds = (unsigned short*)(smem + 32768);  // 128x128 bf16 (32 KB)
  float*          Slds = (float*)(smem + 65536);           // FINAL: 128x128 f32 (64 KB)

  const int tid  = threadIdx.x;
  const int base = blockIdx.x * 128;

  // B = fc_w^T for D=A*B; fc_w is [co][ci] row-major which IS the B-lane
  // layout (column co = contiguous ci run) -> straight bf16 convert.
  for (int idx = tid; idx < CCH*CCH; idx += 256) Blds[idx] = f2bf(fcw[idx]);

  // A tile: 128 tokens x 128 ch with channel-group axial shift (zero pad).
  // groups of ceil(128/5)=26 channels, shifts -2..+2 along AXIS.
  for (int item = tid; item < 128*5; item += 256){
    int tok = item / 5, g = item % 5, s = g - 2;
    int n = base + tok;
    int d = n % HWD, w = (n / HWD) % HWD, h = n / (HWD*HWD);
    int src = n; bool ok;
    if (AXIS == 0){ int hs = h - s; ok = (unsigned)hs < (unsigned)HWD; src = n - s*HWD*HWD; }
    else if (AXIS == 1){ int ws = w - s; ok = (unsigned)ws < (unsigned)HWD; src = n - s*HWD; }
    else { int ds = d - s; ok = (unsigned)ds < (unsigned)HWD; src = n - s; }
    int c0 = g * 26;
    int c1 = c0 + 26; if (c1 > CCH) c1 = CCH;
    const unsigned int* sp = (const unsigned int*)(hin + (long)src*CCH + c0);
    unsigned int* dp = (unsigned int*)(Alds + tok*CCH + c0);
    int cnt = (c1 - c0) >> 1;
    for (int j = 0; j < cnt; ++j) dp[j] = ok ? sp[j] : 0u;
  }
  __syncthreads();

  const int wave = tid >> 5, lane = tid & 31;
  const int mbase = wave * 16;
  const int lh = lane & 15, hi16 = lane >> 4;

  v8f acc[8];
  v8f zero = {0.f,0.f,0.f,0.f,0.f,0.f,0.f,0.f};
  #pragma unroll
  for (int nt = 0; nt < 8; ++nt) acc[nt] = zero;

  #pragma unroll
  for (int kc = 0; kc < 4; ++kc){
    const int k0 = kc * 32;
    // A frag (ISA 7.12.2): lane<16 -> row=lane, K{k0..k0+7, k0+16..k0+23};
    //                      lane>=16 -> row=lane-16, K shifted by +8.
    Frag a;
    const unsigned short* ap = Alds + (mbase + lh)*CCH + k0 + (hi16 ? 8 : 0);
    a.q.lo = *(const u32x4*)(ap);
    a.q.hi = *(const u32x4*)(ap + 16);
    #pragma unroll
    for (int nt = 0; nt < 8; ++nt){
      // B frag: lane<16 -> col=lane, K{k0..k0+7, k0+16..k0+23}; lane>=16 -> K shifted by +8.
      Frag b;
      const unsigned short* bp = Blds + (nt*16 + lh)*CCH + k0 + (hi16 ? 8 : 0);
      b.q.lo = *(const u32x4*)(bp);
      b.q.hi = *(const u32x4*)(bp + 16);
      acc[nt] = __builtin_amdgcn_wmma_f32_16x16x32_bf16(false, a.v, false, b.v,
                                                        (short)0, acc[nt], false, false);
    }
  }

  if (!FINAL){
    #pragma unroll
    for (int nt = 0; nt < 8; ++nt){
      const int co = nt*16 + lh;
      const float bias = fcb[co];
      #pragma unroll
      for (int r = 0; r < 8; ++r){
        int m = hi16 ? (8 + r) : r;
        long gtok = base + mbase + m;
        hout[gtok*CCH + co] = f2bf(acc[nt][r] + bias);
      }
    }
  } else {
    // bias + exact GeLU + residual add
    #pragma unroll
    for (int nt = 0; nt < 8; ++nt){
      const int co = nt*16 + lh;
      const float bias = fcb[co];
      #pragma unroll
      for (int r = 0; r < 8; ++r){
        int m = hi16 ? (8 + r) : r;
        long gtok = base + mbase + m;
        float v = acc[nt][r] + bias;
        v = 0.5f * v * (1.f + erff(v * 0.70710678118654752f));
        acc[nt][r] = v + tres[gtok*CCH + co];
      }
    }
    // final LN: token m's 128 channels live in one lane-half (16 lanes x 8 tiles)
    float sum[8], sq[8];
    #pragma unroll
    for (int r = 0; r < 8; ++r){ sum[r] = 0.f; sq[r] = 0.f; }
    #pragma unroll
    for (int nt = 0; nt < 8; ++nt)
      #pragma unroll
      for (int r = 0; r < 8; ++r){ float v = acc[nt][r]; sum[r] += v; sq[r] += v*v; }
    #pragma unroll
    for (int mask = 1; mask <= 8; mask <<= 1){
      #pragma unroll
      for (int r = 0; r < 8; ++r){
        sum[r] += __shfl_xor(sum[r], mask, 32);
        sq[r]  += __shfl_xor(sq[r],  mask, 32);
      }
    }
    #pragma unroll
    for (int nt = 0; nt < 8; ++nt){
      const int co = nt*16 + lh;
      const float gv = gn[co], bv = bn[co];
      #pragma unroll
      for (int r = 0; r < 8; ++r){
        float mean = sum[r] * (1.f/128.f);
        float var  = sq[r]  * (1.f/128.f) - mean*mean;
        float inv  = rsqrtf(var + EPSF);
        int m = hi16 ? (8 + r) : r;
        Slds[co*128 + mbase + m] = (acc[nt][r] - mean) * inv * gv + bv;
      }
    }
    __syncthreads();
    // coalesced channel-major writeout: out[co][token]
    for (int idx = tid; idx < CCH*128; idx += 256){
      int co = idx >> 7, i = idx & 127;
      outp[(long)co*NTOK + base + i] = Slds[co*128 + i];
    }
  }
}

// ---------------------------------------------------------------------------
// Kernel 3/5: depthwise 3x3x3 conv over 48^3, bf16 in/out, f32 accumulate
// one wave per token, 4 channels per lane; weights staged in LDS
// ---------------------------------------------------------------------------
__global__ void dwconv_kernel(const unsigned short* __restrict__ in,
    const float* __restrict__ w, const float* __restrict__ b,
    unsigned short* __restrict__ out)
{
  __shared__ float wl[CCH*27];
  __shared__ float bl[CCH];
  const int tid = threadIdx.x;
  for (int idx = tid; idx < CCH*27; idx += 256) wl[idx] = w[idx];
  if (tid < CCH) bl[tid] = b[tid];
  __syncthreads();
  const int tok = blockIdx.x*8 + (tid >> 5);
  const int lane = tid & 31;
  const int c0 = lane * 4;
  const int d0 = tok % HWD, w0 = (tok / HWD) % HWD, h0 = tok / (HWD*HWD);
  float acc0 = bl[c0], acc1 = bl[c0+1], acc2 = bl[c0+2], acc3 = bl[c0+3];
  #pragma unroll
  for (int k0 = 0; k0 < 3; ++k0){
    int ih = h0 + k0 - 1; if ((unsigned)ih >= (unsigned)HWD) continue;
    #pragma unroll
    for (int k1 = 0; k1 < 3; ++k1){
      int iw = w0 + k1 - 1; if ((unsigned)iw >= (unsigned)HWD) continue;
      #pragma unroll
      for (int k2 = 0; k2 < 3; ++k2){
        int id = d0 + k2 - 1; if ((unsigned)id >= (unsigned)HWD) continue;
        long nn = ((long)ih*HWD + iw)*HWD + id;
        u32x2 raw = *(const u32x2*)(in + nn*CCH + c0);
        int kk = (k0*3 + k1)*3 + k2;
        acc0 += bf2f((unsigned short)(raw.x & 0xFFFFu)) * wl[(c0+0)*27 + kk];
        acc1 += bf2f((unsigned short)(raw.x >> 16))     * wl[(c0+1)*27 + kk];
        acc2 += bf2f((unsigned short)(raw.y & 0xFFFFu)) * wl[(c0+2)*27 + kk];
        acc3 += bf2f((unsigned short)(raw.y >> 16))     * wl[(c0+3)*27 + kk];
      }
    }
  }
  u32x2 o;
  o.x = (unsigned)f2bf(acc0) | ((unsigned)f2bf(acc1) << 16);
  o.y = (unsigned)f2bf(acc2) | ((unsigned)f2bf(acc3) << 16);
  *(u32x2*)(out + (long)tok*CCH + c0) = o;
}

// ---------------------------------------------------------------------------
extern "C" void kernel_launch(void* const* d_in, const int* in_sizes, int n_in,
                              void* d_out, int out_size, void* d_ws, size_t ws_size,
                              hipStream_t stream)
{
  const float* x    = (const float*)d_in[0];
  const float* pw   = (const float*)d_in[1];
  const float* pb   = (const float*)d_in[2];
  const float* g0   = (const float*)d_in[3];
  const float* b0   = (const float*)d_in[4];
  const float* g2   = (const float*)d_in[5];
  const float* b2   = (const float*)d_in[6];
  const float* fc1w = (const float*)d_in[7];
  const float* fc1b = (const float*)d_in[8];
  const float* dw1w = (const float*)d_in[9];
  const float* dw1b = (const float*)d_in[10];
  const float* fc2w = (const float*)d_in[11];
  const float* fc2b = (const float*)d_in[12];
  const float* dw2w = (const float*)d_in[13];
  const float* dw2b = (const float*)d_in[14];
  const float* fc3w = (const float*)d_in[15];
  const float* fc3b = (const float*)d_in[16];
  const float* gn   = (const float*)d_in[17];
  const float* bn   = (const float*)d_in[18];

  float* t_f32 = (float*)d_ws;                                        // 56.6 MB f32 residual
  unsigned short* bufA = (unsigned short*)((char*)d_ws + (size_t)NTOK*CCH*4);
  unsigned short* bufB = bufA + (size_t)NTOK*CCH;                     // 2 x 28.3 MB bf16 ping-pong
  float* outp = (float*)d_out;

  proj_ln_kernel<<<NTOK, 128, 0, stream>>>(x, pw, pb, g0, b0, g2, b2, t_f32, bufA);
  shift_gemm_kernel<0,false><<<NTOK/128, 256, 65536, stream>>>(bufA, fc1w, fc1b, bufB,
                                                               nullptr, nullptr, nullptr, nullptr);
  dwconv_kernel<<<NTOK/8, 256, 0, stream>>>(bufB, dw1w, dw1b, bufA);
  shift_gemm_kernel<1,false><<<NTOK/128, 256, 65536, stream>>>(bufA, fc2w, fc2b, bufB,
                                                               nullptr, nullptr, nullptr, nullptr);
  dwconv_kernel<<<NTOK/8, 256, 0, stream>>>(bufB, dw2w, dw2b, bufA);
  shift_gemm_kernel<2,true><<<NTOK/128, 256, 131072, stream>>>(bufA, fc3w, fc3b, nullptr,
                                                               t_f32, gn, bn, outp);
}
